// BiLstmCrf_15298673508963
// MI455X (gfx1250) — matrix operand order repaired and checked by reading, not compiled, not run on hardware
//
#include <hip/hip_runtime.h>
#include <hip/hip_bf16.h>

// Problem constants (from reference)
#define LQ   256   // sequence length
#define BQ   64    // batch
#define EQ   256   // embedding dim
#define HQ   256   // per-direction hidden
#define G4H  1024  // 4*H gate width
#define TQ   16    // tags
#define LB   16384 // L*B

typedef __attribute__((ext_vector_type(16))) __bf16 v16bf;
typedef __attribute__((ext_vector_type(8)))  float  v8f;
typedef __attribute__((ext_vector_type(8)))  __bf16 b8vec;

// ---------------------------------------------------------------------------
// WMMA fragment loaders (CDNA5 wave32 layouts, CDNA5 ISA §7.12.2)
// A: 16x32 bf16, lane l holds row M=l&15; K-base = (l>>4)*8, elements K..K+7
//    in VGPR0-3 and K+16..K+23 in VGPR4-7  -> two 16B contiguous loads.
// B: 32x16 bf16 (B[k][n] = W[n][k] for W row-major [N,K]): lane l holds column
//    N=l&15, K-base = (l>>4)*8, K..K+7 and K+16..K+23 -> two 16B loads from W row.
// C/D: lane l holds N=l&15, VGPR r holds M = r + (l>>4)*8.
// ---------------------------------------------------------------------------
__device__ __forceinline__ v16bf load_a_frag(const __bf16* __restrict__ X, int ld, int m0, int k0) {
  const int l = threadIdx.x & 31;
  const __bf16* p = X + (size_t)(m0 + (l & 15)) * ld + (size_t)(k0 + ((l >> 4) << 3));
  b8vec lo = *(const b8vec*)p;
  b8vec hi = *(const b8vec*)(p + 16);
  v16bf a;
#pragma unroll
  for (int i = 0; i < 8; ++i) { a[i] = lo[i]; a[i + 8] = hi[i]; }
  return a;
}

__device__ __forceinline__ v16bf load_bT_frag(const __bf16* W, int ld, int n0, int k0) {
  const int l = threadIdx.x & 31;
  const __bf16* p = W + (size_t)(n0 + (l & 15)) * ld + (size_t)(k0 + ((l >> 4) << 3));
  b8vec lo = *(const b8vec*)p;
  b8vec hi = *(const b8vec*)(p + 16);
  v16bf b;
#pragma unroll
  for (int i = 0; i < 8; ++i) { b[i] = lo[i]; b[i + 8] = hi[i]; }
  return b;
}

__device__ __forceinline__ float sigm(float x) { return 1.0f / (1.0f + __expf(-x)); }

// ---------------------------------------------------------------------------
// Elementwise helpers
// ---------------------------------------------------------------------------
__global__ void cvt_bf16_kernel(const float* __restrict__ s, __bf16* __restrict__ d, int n) {
  int i = blockIdx.x * blockDim.x + threadIdx.x;
  if (i < n) d[i] = (__bf16)s[i];
}

// x0[(t*B+b)*E + e] = bf16(emb[tokens[t*B+b] * E + e])   (pad token 0 -> emb[0], as ref)
__global__ void embed_kernel(const int* __restrict__ tokens, const float* __restrict__ emb,
                             __bf16* __restrict__ x) {
  int i = blockIdx.x * blockDim.x + threadIdx.x;   // over LB*E (exact grid)
  int e  = i & (EQ - 1);
  int tb = i >> 8;                                  // EQ == 256
  int tok = tokens[tb];
  x[i] = (__bf16)emb[(size_t)tok * EQ + e];
}

// dst[t] = src[ rev(t) ] with rev(t) = (t < len[b]) ? len[b]-1-t : t   (reverse_padded)
__global__ void reverse_bf16_kernel(const __bf16* __restrict__ src, __bf16* __restrict__ dst,
                                    const int* __restrict__ lengths, int D) {
  long i = (long)blockIdx.x * blockDim.x + threadIdx.x;   // over LB*D (exact grid)
  int d  = (int)(i % D);
  long tb = i / D;
  int b = (int)(tb % BQ);
  int t = (int)(tb / BQ);
  int len = lengths[b];
  int rt = (t < len) ? (len - 1 - t) : t;
  dst[i] = src[((size_t)rt * BQ + b) * D + d];
}

// xcat[t,b,0:256] = mask * ysf[t,b,:];  xcat[t,b,256:512] = mask * ysb_rev[rev(t),b,:]
__global__ void build_concat_kernel(const float* __restrict__ ysf, const float* __restrict__ ysb_rev,
                                    const int* __restrict__ tokens, const int* __restrict__ lengths,
                                    __bf16* __restrict__ xcat) {
  long i = (long)blockIdx.x * blockDim.x + threadIdx.x;   // over LB*512 (exact grid)
  int d  = (int)(i & 511);
  long tb = i >> 9;
  int b = (int)(tb % BQ);
  int t = (int)(tb / BQ);
  bool m = tokens[t * BQ + b] != 0;
  float v;
  if (d < HQ) {
    v = ysf[((size_t)t * BQ + b) * HQ + d];
  } else {
    int len = lengths[b];
    int rt = (t < len) ? (len - 1 - t) : t;
    v = ysb_rev[((size_t)rt * BQ + b) * HQ + (d - HQ)];
  }
  xcat[i] = (__bf16)(m ? v : 0.0f);
}

// ---------------------------------------------------------------------------
// Batched pregate / emission GEMM: C[M,N] = X[M,K] @ W[N,K]^T + b0 (+ b1)
// block = 256 thr (8 waves); wave -> 16 rows x (NT*16) cols, A frag reused NT times.
// grid = (N/(NT*16), M/128). M=16384, K in {256,512}, N in {1024,16}.
// ---------------------------------------------------------------------------
template <int NT>
__global__ void gemm_bias_kernel(const __bf16* __restrict__ X, int K,
                                 const __bf16* __restrict__ W, int N,
                                 const float* __restrict__ b0, const float* __restrict__ b1,
                                 float* __restrict__ C) {
  const int l  = threadIdx.x & 31;
  const int wv = threadIdx.x >> 5;
  const int m0 = (blockIdx.y * 8 + wv) * 16;
  const int n0 = blockIdx.x * (NT * 16);
  const int ln = l & 15;
  const int lg = l >> 4;

  v8f acc[NT];
#pragma unroll
  for (int j = 0; j < NT; ++j) {
    int n = n0 + j * 16 + ln;
    float bv = b0[n] + (b1 ? b1[n] : 0.0f);
#pragma unroll
    for (int r = 0; r < 8; ++r) acc[j][r] = bv;
  }

  for (int k = 0; k < K; k += 32) {
    v16bf a = load_a_frag(X, K, m0, k);
#pragma unroll
    for (int j = 0; j < NT; ++j) {
      v16bf b = load_bT_frag(W, K, n0 + j * 16, k);
      acc[j] = __builtin_amdgcn_wmma_f32_16x16x32_bf16(false, a, false, b,
                                                       (short)0, acc[j], false, false);
    }
  }

#pragma unroll
  for (int j = 0; j < NT; ++j) {
    int n = n0 + j * 16 + ln;
#pragma unroll
    for (int r = 0; r < 8; ++r) {
      int m = m0 + (lg << 3) + r;
      C[(size_t)m * N + n] = acc[j][r];
    }
  }
}

// ---------------------------------------------------------------------------
// Recurrent LSTM kernel: one 512-thread (16-wave) workgroup per direction.
// Per step: gates[64,1024] = pregate[t] + h_{t-1} @ Whh^T  (WMMA, h in LDS as bf16),
// then LSTM pointwise update; c-state lives in registers across all 256 steps.
// Wave wv owns hidden-column tile ht = wv (16 columns) for all 4 batch tiles
// (units u = mt = 0..3); each unit accumulates its (i,f,g,o) gate quadruple.
// Whh is streamed from L2 each step. An opaque per-iteration *offset* (always 0)
// defeats LICM of the loop-invariant B-fragments (which previously spilled to
// scratch) while preserving the kernel-arg pointer provenance so the loads
// lower to global_load_b128 (not flat_load, which would also tie up DScnt and
// the LDS pipe that the h-state A-fragments need).
// ---------------------------------------------------------------------------
#define HLD 264  // LDS row pitch in bf16 (256 + 8 pad, keeps 16B alignment)

__global__ void lstm_rec_kernel(const float* __restrict__ pgA, const __bf16* __restrict__ WA,
                                float* __restrict__ ysA,
                                const float* __restrict__ pgB, const __bf16* __restrict__ WB,
                                float* __restrict__ ysB) {
  const float*  pg  = blockIdx.x ? pgB : pgA;
  const __bf16* Whh = blockIdx.x ? WB : WA;
  float*        ys  = blockIdx.x ? ysB : ysA;

  __shared__ __bf16 hbuf[BQ * HLD];
  for (int i = threadIdx.x; i < BQ * HLD; i += blockDim.x) hbuf[i] = (__bf16)0.0f;
  __syncthreads();

  const int l  = threadIdx.x & 31;
  const int wv = threadIdx.x >> 5;   // 0..15 == hidden-column tile ht
  const int ln = l & 15;
  const int lg = l >> 4;
  const int hcol = wv * 16 + ln;     // this wave's hidden column (per lane)

  float cst[4][8];
#pragma unroll
  for (int u = 0; u < 4; ++u)
#pragma unroll
    for (int r = 0; r < 8; ++r) cst[u][r] = 0.0f;

  for (int t = 0; t < LQ; ++t) {
    // Opaque zero offset: keeps Whh B-fragment loads inside the time loop
    // without losing the global address space of the underlying pointer.
    size_t toff = 0;
    asm volatile("" : "+s"(toff));
    const __bf16* Wt = Whh + toff;

    const float* pgt = pg + (size_t)t * BQ * G4H;
    float hnew[4][8];

#pragma unroll
    for (int u = 0; u < 4; ++u) {          // u == batch tile mt
      const int m0 = u * 16;
      const int mbase = m0 + (lg << 3);

      v8f acc[4];
#pragma unroll
      for (int g = 0; g < 4; ++g) {
        const int n = g * HQ + hcol;
#pragma unroll
        for (int r = 0; r < 8; ++r) acc[g][r] = pgt[(size_t)(mbase + r) * G4H + n];
      }

#pragma unroll
      for (int k = 0; k < HQ; k += 32) {
        // A fragment: h_{t-1} tile from LDS (ds_load_b128)
        const __bf16* hp = &hbuf[(m0 + ln) * HLD + k + (lg << 3)];
        b8vec lo = *(const b8vec*)hp;
        b8vec hi = *(const b8vec*)(hp + 16);
        v16bf a;
#pragma unroll
        for (int i = 0; i < 8; ++i) { a[i] = lo[i]; a[i + 8] = hi[i]; }
#pragma unroll
        for (int g = 0; g < 4; ++g) {
          v16bf b = load_bT_frag(Wt, HQ, g * HQ + wv * 16, k);
          acc[g] = __builtin_amdgcn_wmma_f32_16x16x32_bf16(false, a, false, b,
                                                           (short)0, acc[g], false, false);
        }
      }

      // LSTM pointwise: i,f,g,o in PyTorch order
#pragma unroll
      for (int r = 0; r < 8; ++r) {
        float ig = sigm(acc[0][r]);
        float fg = sigm(acc[1][r]);
        float gg = tanhf(acc[2][r]);
        float og = sigm(acc[3][r]);
        float c  = fg * cst[u][r] + ig * gg;
        float h  = og * tanhf(c);
        cst[u][r]  = c;
        hnew[u][r] = h;
        ys[((size_t)t * BQ + mbase + r) * HQ + hcol] = h;
      }
    }

    __syncthreads();  // all reads of h_{t-1} complete
#pragma unroll
    for (int u = 0; u < 4; ++u)
#pragma unroll
      for (int r = 0; r < 8; ++r)
        hbuf[(u * 16 + (lg << 3) + r) * HLD + hcol] = (__bf16)hnew[u][r];
    __syncthreads();  // h_t visible to all waves
  }
}

// ---------------------------------------------------------------------------
// Viterbi: lane = tag (T=16), one wave per batch element; cross-tag max via
// __shfl over 16 lanes; sequential backtrack on lane 0.
// ---------------------------------------------------------------------------
__global__ void viterbi_kernel(const float* __restrict__ emis, const int* __restrict__ tokens,
                               const float* __restrict__ start_trans,
                               const float* __restrict__ end_trans,
                               const float* __restrict__ trans,
                               signed char* __restrict__ hist,
                               float* __restrict__ best_out, int* __restrict__ tags_out) {
  const int wv = threadIdx.x >> 5;
  const int b  = blockIdx.x * (blockDim.x >> 5) + wv;
  const int l  = threadIdx.x & 31;
  const int n  = l & 15;   // lanes 16..31 mirror 0..15 (harmless duplicates)

  float score = start_trans[n] + emis[(size_t)b * TQ + n];

  for (int t = 1; t < LQ; ++t) {
    float em = emis[((size_t)t * BQ + b) * TQ + n];
    bool  m  = tokens[t * BQ + b] != 0;
    float best = -3.0e38f; int bi = 0;
#pragma unroll
    for (int p = 0; p < TQ; ++p) {
      float sp = __shfl(score, p, 32) + trans[p * TQ + n];
      if (sp > best) { best = sp; bi = p; }   // strict > : first max wins (argmax)
    }
    if (l < TQ) hist[((size_t)(t - 1) * BQ + b) * TQ + n] = (signed char)bi;
    score = m ? (best + em) : score;
  }

  score += end_trans[n];
  float bestv = -3.0e38f; int bt = 0;
#pragma unroll
  for (int p = 0; p < TQ; ++p) {
    float sp = __shfl(score, p, 32);
    if (sp > bestv) { bestv = sp; bt = p; }
  }

  if (l == 0) {
    best_out[b] = bestv;
    int cur = bt;
    bool mlast = tokens[(LQ - 1) * BQ + b] != 0;
    tags_out[(LQ - 1) * BQ + b] = mlast ? cur : 0;
    for (int t = LQ - 2; t >= 0; --t) {
      int  prev  = hist[((size_t)t * BQ + b) * TQ + cur];
      bool mnext = tokens[(t + 1) * BQ + b] != 0;
      cur = mnext ? prev : cur;
      bool mt = tokens[t * BQ + b] != 0;
      tags_out[t * BQ + b] = mt ? cur : 0;
    }
  }
}

// ---------------------------------------------------------------------------
// Host orchestration
// ---------------------------------------------------------------------------
extern "C" void kernel_launch(void* const* d_in, const int* in_sizes, int n_in,
                              void* d_out, int out_size, void* d_ws, size_t ws_size,
                              hipStream_t stream) {
  (void)in_sizes; (void)n_in; (void)out_size; (void)ws_size;

  const int*   tokens  = (const int*)d_in[0];
  const int*   lengths = (const int*)d_in[1];
  const float* emb     = (const float*)d_in[2];
  const float* w_ih0f = (const float*)d_in[3];
  const float* w_hh0f = (const float*)d_in[4];
  const float* b_ih0f = (const float*)d_in[5];
  const float* b_hh0f = (const float*)d_in[6];
  const float* w_ih0b = (const float*)d_in[7];
  const float* w_hh0b = (const float*)d_in[8];
  const float* b_ih0b = (const float*)d_in[9];
  const float* b_hh0b = (const float*)d_in[10];
  const float* w_ih1f = (const float*)d_in[11];
  const float* w_hh1f = (const float*)d_in[12];
  const float* b_ih1f = (const float*)d_in[13];
  const float* b_hh1f = (const float*)d_in[14];
  const float* w_ih1b = (const float*)d_in[15];
  const float* w_hh1b = (const float*)d_in[16];
  const float* b_ih1b = (const float*)d_in[17];
  const float* b_hh1b = (const float*)d_in[18];
  const float* w_out  = (const float*)d_in[19];
  const float* b_out  = (const float*)d_in[20];
  const float* s_tr   = (const float*)d_in[21];
  const float* e_tr   = (const float*)d_in[22];
  const float* trans  = (const float*)d_in[23];

  char* ws = (char*)d_ws;
  size_t off = 0;
  auto take = [&](size_t bytes) -> char* {
    char* p = ws + off;
    off = (off + bytes + 255) & ~(size_t)255;
    return p;
  };

  // bf16 weights
  __bf16* wb_ih0f = (__bf16*)take((size_t)G4H * EQ * 2);
  __bf16* wb_hh0f = (__bf16*)take((size_t)G4H * HQ * 2);
  __bf16* wb_ih0b = (__bf16*)take((size_t)G4H * EQ * 2);
  __bf16* wb_hh0b = (__bf16*)take((size_t)G4H * HQ * 2);
  __bf16* wb_ih1f = (__bf16*)take((size_t)G4H * 2 * HQ * 2);
  __bf16* wb_hh1f = (__bf16*)take((size_t)G4H * HQ * 2);
  __bf16* wb_ih1b = (__bf16*)take((size_t)G4H * 2 * HQ * 2);
  __bf16* wb_hh1b = (__bf16*)take((size_t)G4H * HQ * 2);
  __bf16* wb_out  = (__bf16*)take((size_t)TQ * 2 * HQ * 2);
  // activations
  __bf16* x0   = (__bf16*)take((size_t)LB * EQ * 2);
  __bf16* x0r  = (__bf16*)take((size_t)LB * EQ * 2);
  __bf16* x1   = (__bf16*)take((size_t)LB * 2 * HQ * 2);   // concat input / emission input
  __bf16* x1r  = (__bf16*)take((size_t)LB * 2 * HQ * 2);
  float*  pgf  = (float*)take((size_t)LB * G4H * 4);
  float*  pgb  = (float*)take((size_t)LB * G4H * 4);
  float*  ysf  = (float*)take((size_t)LB * HQ * 4);
  float*  ysb  = (float*)take((size_t)LB * HQ * 4);
  float*  emis = (float*)take((size_t)LB * TQ * 4);
  signed char* hist = (signed char*)take((size_t)(LQ - 1) * BQ * TQ);

  float* best_out = (float*)d_out;
  int*   tags_out = ((int*)d_out) + BQ;

  // ---- weight conversion f32 -> bf16 ----
  auto cvt = [&](const float* s, __bf16* d, int n) {
    cvt_bf16_kernel<<<(n + 255) / 256, 256, 0, stream>>>(s, d, n);
  };
  cvt(w_ih0f, wb_ih0f, G4H * EQ);
  cvt(w_hh0f, wb_hh0f, G4H * HQ);
  cvt(w_ih0b, wb_ih0b, G4H * EQ);
  cvt(w_hh0b, wb_hh0b, G4H * HQ);
  cvt(w_ih1f, wb_ih1f, G4H * 2 * HQ);
  cvt(w_hh1f, wb_hh1f, G4H * HQ);
  cvt(w_ih1b, wb_ih1b, G4H * 2 * HQ);
  cvt(w_hh1b, wb_hh1b, G4H * HQ);
  cvt(w_out,  wb_out,  TQ * 2 * HQ);

  // ---- embedding + per-sequence reversal ----
  embed_kernel<<<(LB * EQ) / 256, 256, 0, stream>>>(tokens, emb, x0);
  reverse_bf16_kernel<<<((size_t)LB * EQ) / 256, 256, 0, stream>>>(x0, x0r, lengths, EQ);

  // ---- layer 0: pregates (big WMMA GEMMs), then recurrence ----
  gemm_bias_kernel<4><<<dim3(G4H / 64, LB / 128), 256, 0, stream>>>(
      x0, EQ, wb_ih0f, G4H, b_ih0f, b_hh0f, pgf);
  gemm_bias_kernel<4><<<dim3(G4H / 64, LB / 128), 256, 0, stream>>>(
      x0r, EQ, wb_ih0b, G4H, b_ih0b, b_hh0b, pgb);
  lstm_rec_kernel<<<2, 512, 0, stream>>>(pgf, wb_hh0f, ysf, pgb, wb_hh0b, ysb);

  // ---- concat (masked) + reversal for layer 1 ----
  build_concat_kernel<<<((size_t)LB * 2 * HQ) / 256, 256, 0, stream>>>(
      ysf, ysb, tokens, lengths, x1);
  reverse_bf16_kernel<<<((size_t)LB * 2 * HQ) / 256, 256, 0, stream>>>(x1, x1r, lengths, 2 * HQ);

  // ---- layer 1 ----
  gemm_bias_kernel<4><<<dim3(G4H / 64, LB / 128), 256, 0, stream>>>(
      x1, 2 * HQ, wb_ih1f, G4H, b_ih1f, b_hh1f, pgf);
  gemm_bias_kernel<4><<<dim3(G4H / 64, LB / 128), 256, 0, stream>>>(
      x1r, 2 * HQ, wb_ih1b, G4H, b_ih1b, b_hh1b, pgb);
  lstm_rec_kernel<<<2, 512, 0, stream>>>(pgf, wb_hh1f, ysf, pgb, wb_hh1b, ysb);

  // ---- emissions = concat @ w_out^T + b_out ----
  build_concat_kernel<<<((size_t)LB * 2 * HQ) / 256, 256, 0, stream>>>(
      ysf, ysb, tokens, lengths, x1);
  gemm_bias_kernel<1><<<dim3(1, LB / 128), 256, 0, stream>>>(
      x1, 2 * HQ, wb_out, TQ, b_out, (const float*)nullptr, emis);

  // ---- Viterbi decode ----
  viterbi_kernel<<<2, 1024, 0, stream>>>(emis, tokens, s_tr, e_tr, trans,
                                         hist, best_out, tags_out);
}
